// highwayNet_5351529251237
// MI455X (gfx1250) — matrix operand order repaired and checked by reading, not compiled, not run on hardware
//
#include <hip/hip_runtime.h>
#include <stdint.h>

// ---------------- problem constants ----------------
#define BATCH   128
#define HN      39
#define WN      39
#define PIX     (HN*WN)             // 1521
#define NNODES  (BATCH*WN)          // 4992
#define EPER    1563
#define EBASE   (BATCH*EPER)        // 200064
#define ETOT    (EBASE + NNODES)    // 205056 (edges + self loops)
#define HEADS   8
#define HID     64
#define FD1     (HEADS*HID)         // 512
#define K1IN    78
#define K1PAD   96                  // pad 78 -> 3 * 32
#define M1      (2*FD1)             // 1024 (Wl || Wr)
#define K2PAD   512
#define M2      128                 // 2 * 64 (Wl || Wr)
#define EPSBN   1e-5f
#define NEGBIG  -3.0e38f

typedef __attribute__((ext_vector_type(16))) _Float16 v16h;
typedef __attribute__((ext_vector_type(8)))  float    v8f;

// ---------------- helpers ----------------
__device__ __forceinline__ void atomicMaxF(float* addr, float val) {
  if (val >= 0.0f) atomicMax((int*)addr, __float_as_int(val));
  else             atomicMin((unsigned int*)addr, __float_as_uint(val));
}

__global__ void k_fill(float* __restrict__ p, float v, int n) {
  int i = blockIdx.x * blockDim.x + threadIdx.x;
  if (i < n) p[i] = v;
}

// ---------------- conv1 (1x1, 3->8) + BN + ReLU ----------------
__global__ void k_conv1(const float* __restrict__ ve, const float* __restrict__ ac,
                        const float* __restrict__ man,
                        const float* __restrict__ w1, const float* __restrict__ b1,
                        const float* __restrict__ g1, const float* __restrict__ be1,
                        float* __restrict__ x1) {
  int t = blockIdx.x * blockDim.x + threadIdx.x;
  if (t >= BATCH * PIX) return;
  int b  = t / PIX;
  int ij = t - b * PIX;
  float c0 = man[t]; c0 = (c0 == c0) ? c0 : 0.0f;   // NaN -> 0
  float c1 = ac[t];  c1 = (c1 == c1) ? c1 : 0.0f;
  float c2 = ve[t];  c2 = (c2 == c2) ? c2 : 0.0f;
  float inv = rsqrtf(1.0f + EPSBN);
#pragma unroll
  for (int o = 0; o < 8; o++) {
    float y = b1[o] + c0 * w1[o*3+0] + c1 * w1[o*3+1] + c2 * w1[o*3+2];
    y = y * (g1[o] * inv) + be1[o];
    x1[(size_t)(b*8 + o) * PIX + ij] = fmaxf(y, 0.0f);
  }
}

// ---------------- conv2 (3x3 pad1, 8->16) + BN ----------------
__global__ void k_conv2(const float* __restrict__ x1, const float* __restrict__ w2,
                        const float* __restrict__ b2, const float* __restrict__ g2,
                        const float* __restrict__ be2, float* __restrict__ x2) {
  int t = blockIdx.x * blockDim.x + threadIdx.x;
  if (t >= BATCH * PIX) return;
  int b = t / PIX;
  int r = t - b * PIX;
  int i = r / WN, j = r - (r / WN) * WN;
  float inv = rsqrtf(1.0f + EPSBN);
  for (int o = 0; o < 16; o++) {
    float acc = b2[o];
    for (int c = 0; c < 8; c++) {
      const float* xp = x1 + (size_t)(b*8 + c) * PIX;
#pragma unroll
      for (int dy = -1; dy <= 1; dy++) {
        int ii = i + dy;
        if (ii < 0 || ii >= HN) continue;
#pragma unroll
        for (int dx = -1; dx <= 1; dx++) {
          int jj = j + dx;
          if (jj < 0 || jj >= WN) continue;
          acc += xp[ii*WN + jj] * w2[((o*8 + c)*3 + (dy+1))*3 + (dx+1)];
        }
      }
    }
    acc = acc * (g2[o] * inv) + be2[o];
    x2[(size_t)(b*16 + o) * PIX + r] = acc;
  }
}

// ---------------- GRU(16->1) over rows, one thread per (b,w) ----------------
__global__ void k_gru(const float* __restrict__ x2, const float* __restrict__ wih,
                      const float* __restrict__ whh, const float* __restrict__ bih,
                      const float* __restrict__ bhh, float* __restrict__ enc1) {
  int t = blockIdx.x * blockDim.x + threadIdx.x;
  if (t >= NNODES) return;
  int b = t / WN, w = t - (t / WN) * WN;
  float h = 0.0f;
  for (int tt = 0; tt < HN; tt++) {
    float xt[16];
#pragma unroll
    for (int c = 0; c < 16; c++)
      xt[c] = x2[((size_t)(b*16 + c) * HN + tt) * WN + w];
    float gi0 = bih[0], gi1 = bih[1], gi2 = bih[2];
#pragma unroll
    for (int c = 0; c < 16; c++) {
      gi0 += xt[c] * wih[0*16 + c];
      gi1 += xt[c] * wih[1*16 + c];
      gi2 += xt[c] * wih[2*16 + c];
    }
    float gh0 = bhh[0] + h * whh[0];
    float gh1 = bhh[1] + h * whh[1];
    float gh2 = bhh[2] + h * whh[2];
    float rr = 1.0f / (1.0f + expf(-(gi0 + gh0)));
    float zz = 1.0f / (1.0f + expf(-(gi1 + gh1)));
    float nn = tanhf(gi2 + rr * gh2);
    h = (1.0f - zz) * nn + zz * h;
    enc1[((size_t)b * HN + tt) * WN + w] = h;
  }
}

// ---------------- build padded fp16 node features (N x 96) ----------------
__global__ void k_xn(const float* __restrict__ man, const float* __restrict__ mask,
                     const float* __restrict__ enc1, _Float16* __restrict__ xnh) {
  int i = blockIdx.x * blockDim.x + threadIdx.x;
  if (i >= NNODES * K1PAD) return;
  int node = i / K1PAD, f = i - node * K1PAD;
  int b = node / WN, w = node - b * WN;
  float v = 0.0f;
  if (f < HN) {
    float m = man[((size_t)b * HN + f) * WN + w];
    m = (m == m) ? m : 0.0f;
    v = m * mask[b * WN + w];
  } else if (f < 2 * HN) {
    v = enc1[((size_t)b * HN + (f - HN)) * WN + w] * mask[b * WN + w];
  }
  xnh[i] = (_Float16)v;
}

// ---------------- pack (Wl || Wr) -> fp16, K zero-padded; biases concat -----
__global__ void k_pack_w(const float* __restrict__ wl, const float* __restrict__ bl,
                         const float* __restrict__ wr, const float* __restrict__ br,
                         _Float16* __restrict__ wcat, float* __restrict__ bcat,
                         int mhalf, int kin, int kpad) {
  int i = blockIdx.x * blockDim.x + threadIdx.x;
  if (i >= 2 * mhalf * kpad) return;
  int m = i / kpad, k = i - m * kpad;
  float v = 0.0f;
  if (k < kin) v = (m < mhalf) ? wl[(size_t)m * kin + k] : wr[(size_t)(m - mhalf) * kin + k];
  wcat[i] = (_Float16)v;
  if (k == 0) bcat[m] = (m < mhalf) ? bl[m] : br[m - mhalf];
}

// ---------------- edge list: flattened edges + self loops ----------------
__global__ void k_edges(const int* __restrict__ ei, int* __restrict__ es, int* __restrict__ ed) {
  int i = blockIdx.x * blockDim.x + threadIdx.x;
  if (i >= ETOT) return;
  if (i < EBASE) { es[i] = ei[i]; ed[i] = ei[EBASE + i]; }
  else           { es[i] = i - EBASE; ed[i] = i - EBASE; }
}

// ---------------- WMMA GEMM: C(nrows x mcols) = A(nrows x kpad,f16) * W^T + bias
// A row-major fp16, W (mcols x kpad) row-major fp16.
// One wave computes a 16x64 output tile: per k-step one A fragment is reused
// across 4 B fragments / 4 WMMAs (A traffic cut 4x, wmma:vmem ratio up).
__global__ void k_gemm_wmma(const _Float16* __restrict__ A, const _Float16* __restrict__ W,
                            const float* __restrict__ bias, float* __restrict__ C,
                            int nrows, int mcols, int kpad) {
  int wave = (int)((blockIdx.x * blockDim.x + threadIdx.x) >> 5);
  int tilesM = mcols >> 6;                         // 64-wide feature tiles
  int totalTiles = (nrows >> 4) * tilesM;
  if (wave >= totalTiles) return;                  // wave-uniform: EXEC stays full
  int tN = wave / tilesM;                          // node tile (16 rows)
  int tM = wave - tN * tilesM;                     // feature tile (64 cols)
  int lane = threadIdx.x & 31;
  int half = lane >> 4;
  int l16  = lane & 15;
  const uint32_t* arow = (const uint32_t*)(A + (size_t)(tN * 16 + l16) * kpad);
  const uint32_t* wrow[4];
#pragma unroll
  for (int j = 0; j < 4; j++)
    wrow[j] = (const uint32_t*)(W + (size_t)(tM * 64 + j * 16 + l16) * kpad);

  v8f acc0 = {}, acc1 = {}, acc2 = {}, acc3 = {};
  for (int kk = 0; kk < kpad; kk += 32) {
    union { v16h v; uint32_t u[8]; } Af, B0, B1, B2, B3;
    int base = kk >> 1;                            // dword index within row
#pragma unroll
    for (int v = 0; v < 8; v++) {
      // A frag (ISA 7.12.2): lane row = l16; K pair = (v<4?0:16) + 2*(v&3) + 8*half
      int ka = ((v < 4) ? 0 : 16) + ((v & 3) << 1) + (half << 3);
      Af.u[v] = arow[base + (ka >> 1)];
      // B frag: lane col = l16; K pair = same as A
      int kb = base + (ka >> 1);
      B0.u[v] = wrow[0][kb];
      B1.u[v] = wrow[1][kb];
      B2.u[v] = wrow[2][kb];
      B3.u[v] = wrow[3][kb];
    }
    acc0 = __builtin_amdgcn_wmma_f32_16x16x32_f16(false, Af.v, false, B0.v, (short)0, acc0, false, false);
    acc1 = __builtin_amdgcn_wmma_f32_16x16x32_f16(false, Af.v, false, B1.v, (short)0, acc1, false, false);
    acc2 = __builtin_amdgcn_wmma_f32_16x16x32_f16(false, Af.v, false, B2.v, (short)0, acc2, false, false);
    acc3 = __builtin_amdgcn_wmma_f32_16x16x32_f16(false, Af.v, false, B3.v, (short)0, acc3, false, false);
  }

#pragma unroll
  for (int j = 0; j < 4; j++) {
    v8f a = (j == 0) ? acc0 : (j == 1) ? acc1 : (j == 2) ? acc2 : acc3;
    int col = tM * 64 + j * 16 + l16;
    float bv = bias[col];
#pragma unroll
    for (int r = 0; r < 8; r++) {
      int row = tN * 16 + (half << 3) + r;         // D layout: VGPR r -> M = r + 8*half
      C[(size_t)row * mcols + col] = a[r] + bv;
    }
  }
}

// ---------------- GATv2 edge logits + segment max ----------------
__global__ void k_edge_logits(const float* __restrict__ xl, const float* __restrict__ xr,
                              const float* __restrict__ att, const int* __restrict__ es,
                              const int* __restrict__ ed, float* __restrict__ logit,
                              float* __restrict__ mbuf, int heads, int ch, int ldx) {
  int t = blockIdx.x * blockDim.x + threadIdx.x;
  if (t >= ETOT * heads) return;
  int e = t / heads, h = t - e * heads;
  int s = es[e], d = ed[e];
  const float* pl = xl + (size_t)s * ldx + h * ch;
  const float* pr = xr + (size_t)d * ldx + h * ch;
  const float* pa = att + h * ch;
  float acc = 0.0f;
  for (int c = 0; c < ch; c++) {
    float v = pl[c] + pr[c];
    v = (v > 0.0f) ? v : 0.2f * v;                 // leaky_relu(0.2)
    acc += pa[c] * v;
  }
  logit[t] = acc;
  atomicMaxF(&mbuf[d * heads + h], acc);
}

// ---------------- exp(logit - max) + segment sum ----------------
__global__ void k_edge_exp(const float* __restrict__ logit, const int* __restrict__ ed,
                           const float* __restrict__ mbuf, float* __restrict__ den,
                           float* __restrict__ ex, int heads) {
  int t = blockIdx.x * blockDim.x + threadIdx.x;
  if (t >= ETOT * heads) return;
  int e = t / heads, h = t - e * heads;
  int d = ed[e];
  float v = expf(logit[t] - mbuf[d * heads + h]);
  ex[t] = v;
  atomicAdd(&den[d * heads + h], v);
}

// ---------------- out[dst] += xl[src] * alpha ----------------
__global__ void k_edge_acc(const float* __restrict__ xl, const int* __restrict__ es,
                           const int* __restrict__ ed, const float* __restrict__ ex,
                           const float* __restrict__ den, float* __restrict__ out,
                           int heads, int ch, int ldx) {
  int t = blockIdx.x * blockDim.x + threadIdx.x;
  if (t >= ETOT * heads) return;
  int e = t / heads, h = t - e * heads;
  int s = es[e], d = ed[e];
  float alpha = ex[t] / den[d * heads + h];
  const float* pl = xl + (size_t)s * ldx + h * ch;
  float* po = out + (size_t)d * (heads * ch) + h * ch;
  for (int c = 0; c < ch; c++) atomicAdd(&po[c], pl[c] * alpha);
}

// ---------------- bias + ELU -> fp16 for next layer ----------------
__global__ void k_elu_h16(const float* __restrict__ in, const float* __restrict__ bias,
                          _Float16* __restrict__ outh, int n, int fdim) {
  int i = blockIdx.x * blockDim.x + threadIdx.x;
  if (i >= n) return;
  float v = in[i] + bias[i % fdim];
  v = (v > 0.0f) ? v : (expf(v) - 1.0f);
  outh[i] = (_Float16)v;
}

// ---------------- final bias add ----------------
__global__ void k_final(const float* __restrict__ in, const float* __restrict__ bias,
                        float* __restrict__ out, int n, int fdim) {
  int i = blockIdx.x * blockDim.x + threadIdx.x;
  if (i >= n) return;
  out[i] = in[i] + bias[i % fdim];
}

// =======================================================================
extern "C" void kernel_launch(void* const* d_in, const int* in_sizes, int n_in,
                              void* d_out, int out_size, void* d_ws, size_t ws_size,
                              hipStream_t stream) {
  (void)in_sizes; (void)n_in; (void)out_size; (void)ws_size;
  const int*   ei    = (const int*)  d_in[0];   // (2,B,EPER)
  const float* ve    = (const float*)d_in[1];
  const float* ac    = (const float*)d_in[2];
  const float* man   = (const float*)d_in[3];
  const float* mask  = (const float*)d_in[4];
  // d_in[5] graph_matrix: overwritten by reference, unused
  const float* c1w   = (const float*)d_in[6];
  const float* c1b   = (const float*)d_in[7];
  const float* bn1g  = (const float*)d_in[8];
  const float* bn1b  = (const float*)d_in[9];
  const float* c2w   = (const float*)d_in[10];
  const float* c2b   = (const float*)d_in[11];
  const float* bn2g  = (const float*)d_in[12];
  const float* bn2b  = (const float*)d_in[13];
  const float* wih   = (const float*)d_in[14];
  const float* whh   = (const float*)d_in[15];
  const float* bih   = (const float*)d_in[16];
  const float* bhh   = (const float*)d_in[17];
  const float* g1wl  = (const float*)d_in[18];
  const float* g1bl  = (const float*)d_in[19];
  const float* g1wr  = (const float*)d_in[20];
  const float* g1br  = (const float*)d_in[21];
  const float* g1att = (const float*)d_in[22];
  const float* g1bias= (const float*)d_in[23];
  const float* g2wl  = (const float*)d_in[24];
  const float* g2bl  = (const float*)d_in[25];
  const float* g2wr  = (const float*)d_in[26];
  const float* g2br  = (const float*)d_in[27];
  const float* g2att = (const float*)d_in[28];
  const float* g2bias= (const float*)d_in[29];
  float* out = (float*)d_out;

  // ---- workspace carve-up ----
  char* ws = (char*)d_ws;
  size_t off = 0;
  auto alloc = [&](size_t bytes) -> void* {
    off = (off + 255) & ~(size_t)255;
    void* p = ws + off;
    off += bytes;
    return p;
  };
  float*    x1    = (float*)   alloc((size_t)BATCH*8*PIX*4);
  float*    x2    = (float*)   alloc((size_t)BATCH*16*PIX*4);
  float*    enc1  = (float*)   alloc((size_t)BATCH*PIX*4);
  _Float16* xnh   = (_Float16*)alloc((size_t)NNODES*K1PAD*2);
  _Float16* w1h   = (_Float16*)alloc((size_t)M1*K1PAD*2);
  float*    b1cat = (float*)   alloc((size_t)M1*4);
  _Float16* w2h   = (_Float16*)alloc((size_t)M2*K2PAD*2);
  float*    b2cat = (float*)   alloc((size_t)M2*4);
  float*    xlr1  = (float*)   alloc((size_t)NNODES*M1*4);   // [xl | xr] GAT1
  int*      es    = (int*)     alloc((size_t)ETOT*4);
  int*      ed    = (int*)     alloc((size_t)ETOT*4);
  float*    lg1   = (float*)   alloc((size_t)ETOT*HEADS*4);
  float*    ex1   = (float*)   alloc((size_t)ETOT*HEADS*4);
  float*    m1    = (float*)   alloc((size_t)NNODES*HEADS*4);
  float*    den1  = (float*)   alloc((size_t)NNODES*HEADS*4);
  float*    out1  = (float*)   alloc((size_t)NNODES*FD1*4);
  _Float16* h1h   = (_Float16*)alloc((size_t)NNODES*FD1*2);
  float*    xlr2  = (float*)   alloc((size_t)NNODES*M2*4);   // [xl | xr] GAT2
  float*    lg2   = (float*)   alloc((size_t)ETOT*4);
  float*    ex2   = (float*)   alloc((size_t)ETOT*4);
  float*    m2    = (float*)   alloc((size_t)NNODES*4);
  float*    den2  = (float*)   alloc((size_t)NNODES*4);
  float*    out2  = (float*)   alloc((size_t)NNODES*HID*4);

  const int T = 256;
  auto nb = [](long long n, int t) { return (unsigned)((n + t - 1) / t); };

  // ---- init accumulators (ws is poisoned each run) ----
  k_fill<<<nb(NNODES*HEADS,T),T,0,stream>>>(m1,  NEGBIG, NNODES*HEADS);
  k_fill<<<nb(NNODES*HEADS,T),T,0,stream>>>(den1, 0.0f,  NNODES*HEADS);
  k_fill<<<nb((long long)NNODES*FD1,T),T,0,stream>>>(out1, 0.0f, NNODES*FD1);
  k_fill<<<nb(NNODES,T),T,0,stream>>>(m2,  NEGBIG, NNODES);
  k_fill<<<nb(NNODES,T),T,0,stream>>>(den2, 0.0f,  NNODES);
  k_fill<<<nb((long long)NNODES*HID,T),T,0,stream>>>(out2, 0.0f, NNODES*HID);

  // ---- packing / edge list (independent of conv chain) ----
  k_pack_w<<<nb((long long)M1*K1PAD,T),T,0,stream>>>(g1wl, g1bl, g1wr, g1br, w1h, b1cat, FD1, K1IN, K1PAD);
  k_pack_w<<<nb((long long)M2*K2PAD,T),T,0,stream>>>(g2wl, g2bl, g2wr, g2br, w2h, b2cat, HID, FD1, K2PAD);
  k_edges<<<nb(ETOT,T),T,0,stream>>>(ei, es, ed);

  // ---- conv stack + GRU + node features ----
  k_conv1<<<nb((long long)BATCH*PIX,T),T,0,stream>>>(ve, ac, man, c1w, c1b, bn1g, bn1b, x1);
  k_conv2<<<nb((long long)BATCH*PIX,T),T,0,stream>>>(x1, c2w, c2b, bn2g, bn2b, x2);
  k_gru<<<nb(NNODES,T),T,0,stream>>>(x2, wih, whh, bih, bhh, enc1);
  k_xn<<<nb((long long)NNODES*K1PAD,T),T,0,stream>>>(man, mask, enc1, xnh);

  // ---- GAT layer 1 ----
  {
    long long waves = (long long)(NNODES/16) * (M1/64);
    k_gemm_wmma<<<nb(waves*32, 128), 128, 0, stream>>>(xnh, w1h, b1cat, xlr1, NNODES, M1, K1PAD);
  }
  k_edge_logits<<<nb((long long)ETOT*HEADS,T),T,0,stream>>>(xlr1, xlr1 + FD1, g1att, es, ed, lg1, m1, HEADS, HID, M1);
  k_edge_exp  <<<nb((long long)ETOT*HEADS,T),T,0,stream>>>(lg1, ed, m1, den1, ex1, HEADS);
  k_edge_acc  <<<nb((long long)ETOT*HEADS,T),T,0,stream>>>(xlr1, es, ed, ex1, den1, out1, HEADS, HID, M1);
  k_elu_h16   <<<nb((long long)NNODES*FD1,T),T,0,stream>>>(out1, g1bias, h1h, NNODES*FD1, FD1);

  // ---- GAT layer 2 ----
  {
    long long waves = (long long)(NNODES/16) * (M2/64);
    k_gemm_wmma<<<nb(waves*32, 128), 128, 0, stream>>>(h1h, w2h, b2cat, xlr2, NNODES, M2, K2PAD);
  }
  k_edge_logits<<<nb(ETOT,T),T,0,stream>>>(xlr2, xlr2 + HID, g2att, es, ed, lg2, m2, 1, HID, M2);
  k_edge_exp  <<<nb(ETOT,T),T,0,stream>>>(lg2, ed, m2, den2, ex2, 1);
  k_edge_acc  <<<nb(ETOT,T),T,0,stream>>>(xlr2, es, ed, ex2, den2, out2, 1, HID, M2);
  k_final     <<<nb((long long)NNODES*HID,T),T,0,stream>>>(out2, g2bias, out, NNODES*HID, HID);
}
